// PCLLoss_85134841741959
// MI455X (gfx1250) — matrix operand order repaired and checked by reading, not compiled, not run on hardware
//
#include <hip/hip_runtime.h>
#include <hip/hip_bf16.h>

typedef __attribute__((ext_vector_type(16))) _Float16 v16h;
typedef __attribute__((ext_vector_type(8)))  float    v8f;

#define B_ROWS   8192
#define D_FEAT   128
#define NCLUS    256
#define NLVL     3
#define INV_TAU  (1.0f/0.07f)
#define ALPHA_C  10.0f
#define EPS_NORM 1e-12f
#define EPS_DENS 1e-6f
#define NEG_BIG  (-3.0e38f)

// ---------------------------------------------------------------------------
// WMMA fragment loaders (f16, 16x16x32). Row-major source, stride D_FEAT.
// A 16x32: lanes 0-15 & 16-31 both hold M = lane&15; K interleave per ISA:
//   lanes<16 : K = {k0+0..7, k0+16..23},  lanes>=16 : K = {k0+8..15, k0+24..31}
// B 32x16: lane owns column n = lane&15; same K interleave as A:
//   lanes<16 : K = {k0+0..7, k0+16..23},  lanes>=16 : K = {k0+8..15, k0+24..31}
// ---------------------------------------------------------------------------
__device__ inline v16h load_frag_a(const _Float16* __restrict__ fh,
                                   int r0, int k0, int lane) {
  const int hi  = (lane >> 4) & 1;
  const int row = r0 + (lane & 15);
  const _Float16* p = fh + row * D_FEAT + k0 + hi * 8;
  v16h f;
#pragma unroll
  for (int e = 0; e < 8; ++e) { f[e] = p[e]; f[e + 8] = p[e + 16]; }
  return f;
}

__device__ inline v16h load_frag_b(const _Float16* __restrict__ fh,
                                   int c0, int k0, int lane) {
  const int hi  = (lane >> 4) & 1;
  const int col = c0 + (lane & 15);
  const _Float16* p = fh + col * D_FEAT + k0 + hi * 8;
  v16h f;
#pragma unroll
  for (int e = 0; e < 8; ++e) { f[e] = p[e]; f[e + 8] = p[e + 16]; }
  return f;
}

#define WMMA_F16(A, B, C) \
  __builtin_amdgcn_wmma_f32_16x16x32_f16(false, (A), false, (B), (short)0, (C), false, false)

// ---------------------------------------------------------------------------
// Kernel 1: row-normalize features; emit fp32 and f16 copies.
// ---------------------------------------------------------------------------
__global__ __launch_bounds__(256) void k_normalize(
    const float* __restrict__ feat, float* __restrict__ feat_n,
    _Float16* __restrict__ feat_h) {
  const int wave = threadIdx.x >> 5;
  const int lane = threadIdx.x & 31;
  const int row  = blockIdx.x * 8 + wave;
  const float* p = feat + row * D_FEAT + lane * 4;
  float v0 = p[0], v1 = p[1], v2 = p[2], v3 = p[3];
  float ss = v0 * v0 + v1 * v1 + v2 * v2 + v3 * v3;
#pragma unroll
  for (int m = 16; m >= 1; m >>= 1) ss += __shfl_xor(ss, m, 32);
  const float inv = 1.0f / fmaxf(sqrtf(ss), EPS_NORM);
  float*     q = feat_n + row * D_FEAT + lane * 4;
  _Float16*  h = feat_h + row * D_FEAT + lane * 4;
  q[0] = v0 * inv; q[1] = v1 * inv; q[2] = v2 * inv; q[3] = v3 * inv;
  h[0] = (_Float16)(v0 * inv); h[1] = (_Float16)(v1 * inv);
  h[2] = (_Float16)(v2 * inv); h[3] = (_Float16)(v3 * inv);
}

// ---------------------------------------------------------------------------
// Kernel 2: instance loss. One block per 16-row tile (512 blocks, 4 waves).
// Logits are bounded: unit rows => dot<=1 => v<=1/tau. Use FIXED max M=1/tau:
// no online-max, no rescaling; exp(v-M) in [4e-13, ~1], sum <= 8192. This is
// exactly the reference log-softmax shifted by a constant. Labels cached in
// LDS so the inner loop's only VMEM is the B fragments.
// ---------------------------------------------------------------------------
__global__ __launch_bounds__(128) void k_insloss(
    const _Float16* __restrict__ feat_h, const int* __restrict__ labels,
    float* __restrict__ out) {
  __shared__ int   slab[B_ROWS];       // 32 KB: level-0 labels
  __shared__ float red[4][16][3];
  const int wave = threadIdx.x >> 5;
  const int lane = threadIdx.x & 31;
  const int hi   = (lane >> 4) & 1;
  const int r0   = blockIdx.x * 16;

  for (int i = threadIdx.x; i < B_ROWS; i += 128) slab[i] = labels[i * NLVL + 0];
  __syncthreads();

  const v16h a0 = load_frag_a(feat_h, r0,  0, lane);
  const v16h a1 = load_frag_a(feat_h, r0, 32, lane);
  const v16h a2 = load_frag_a(feat_h, r0, 64, lane);
  const v16h a3 = load_frag_a(feat_h, r0, 96, lane);

  int labRow[8];
#pragma unroll
  for (int j = 0; j < 8; ++j) labRow[j] = slab[r0 + j + hi * 8];

  float rsum[8], rpos[8], rcnt[8];
#pragma unroll
  for (int j = 0; j < 8; ++j) { rsum[j] = 0.f; rpos[j] = 0.f; rcnt[j] = 0.f; }

  for (int nt = wave; nt < B_ROWS / 16; nt += 4) {
    const int n0 = nt * 16;
    const v16h b0 = load_frag_b(feat_h, n0,  0, lane);
    const v16h b1 = load_frag_b(feat_h, n0, 32, lane);
    const v16h b2 = load_frag_b(feat_h, n0, 64, lane);
    const v16h b3 = load_frag_b(feat_h, n0, 96, lane);
    v8f acc = {};
    acc = WMMA_F16(a0, b0, acc);
    acc = WMMA_F16(a1, b1, acc);
    acc = WMMA_F16(a2, b2, acc);
    acc = WMMA_F16(a3, b3, acc);
    const int labCol = slab[n0 + (lane & 15)];
#pragma unroll
    for (int j = 0; j < 8; ++j) {
      const float v = acc[j] * INV_TAU;
      rsum[j] += __expf(v - INV_TAU);          // fixed-max softmax term
      const bool match = (labCol == labRow[j]);
      rpos[j] += match ? v : 0.f;
      rcnt[j] += match ? 1.f : 0.f;
    }
  }
  // plain sums across the 16 lanes of each half-wave (masks stay within halves)
#pragma unroll
  for (int j = 0; j < 8; ++j) {
#pragma unroll
    for (int m = 1; m < 16; m <<= 1) {
      rsum[j] += __shfl_xor(rsum[j], m, 32);
      rpos[j] += __shfl_xor(rpos[j], m, 32);
      rcnt[j] += __shfl_xor(rcnt[j], m, 32);
    }
  }
  if ((lane & 15) == 0) {
#pragma unroll
    for (int j = 0; j < 8; ++j) {
      const int m = j + hi * 8;
      red[wave][m][0] = rsum[j]; red[wave][m][1] = rpos[j]; red[wave][m][2] = rcnt[j];
    }
  }
  __syncthreads();
  if (threadIdx.x < 16) {
    const int m = threadIdx.x;
    float gs = 0.f, gp = 0.f, gc = 0.f;
    for (int w = 0; w < 4; ++w) {
      gs += red[w][m][0]; gp += red[w][m][1]; gc += red[w][m][2];
    }
    const float lse = INV_TAU + logf(gs);
    out[r0 + m] = lse - gp / fmaxf(gc, 1.0f);
  }
}

// ---------------------------------------------------------------------------
// Kernel 3: deterministic per-cluster segment sum (prototype numerators +
// sizes). One block per cluster; labels cached in LDS; fixed iteration order.
// ---------------------------------------------------------------------------
__global__ __launch_bounds__(128) void k_proto_accum(
    const float* __restrict__ feat_n, const int* __restrict__ labels, int lvl,
    float* __restrict__ proto, float* __restrict__ sizes) {
  __shared__ int slab[B_ROWS];
  const int c = blockIdx.x;
  const int k = threadIdx.x;
  for (int i = k; i < B_ROWS; i += 128) slab[i] = labels[i * NLVL + lvl];
  __syncthreads();
  float acc = 0.f, cnt = 0.f;
  for (int i = 0; i < B_ROWS; ++i) {
    if (slab[i] == c) { acc += feat_n[i * D_FEAT + k]; cnt += 1.f; }
  }
  proto[c * D_FEAT + k] = acc;
  if (k == 0) sizes[c] = cnt;
}

// ---------------------------------------------------------------------------
// Kernel 4: prototypes = sum / max(size,1) (in place); normalized f16 copy.
// ---------------------------------------------------------------------------
__global__ __launch_bounds__(128) void k_proto_final(
    float* __restrict__ proto, const float* __restrict__ sizes,
    _Float16* __restrict__ proto_h) {
  __shared__ float ss[128];
  const int c = blockIdx.x, k = threadIdx.x;
  const float safe = fmaxf(sizes[c], 1.0f);
  const float p = proto[c * D_FEAT + k] / safe;
  proto[c * D_FEAT + k] = p;
  ss[k] = p * p;
  __syncthreads();
  for (int s = 64; s >= 1; s >>= 1) { if (k < s) ss[k] += ss[k + s]; __syncthreads(); }
  const float inv = 1.0f / fmaxf(sqrtf(ss[0]), EPS_NORM);
  proto_h[c * D_FEAT + k] = (_Float16)(p * inv);
}

// ---------------------------------------------------------------------------
// Kernel 5: per-row distance to its (unnormalized) prototype.
// ---------------------------------------------------------------------------
__global__ __launch_bounds__(256) void k_dist_row(
    const float* __restrict__ feat_n, const float* __restrict__ proto,
    const int* __restrict__ labels, int lvl, float* __restrict__ dist) {
  const int wave = threadIdx.x >> 5;
  const int lane = threadIdx.x & 31;
  const int row  = blockIdx.x * 8 + wave;
  const int lab  = labels[row * NLVL + lvl];
  const float* f = feat_n + row * D_FEAT + lane * 4;
  const float* p = proto  + lab * D_FEAT + lane * 4;
  float ss = 0.f;
#pragma unroll
  for (int e = 0; e < 4; ++e) { const float d = f[e] - p[e]; ss += d * d; }
#pragma unroll
  for (int m = 16; m >= 1; m >>= 1) ss += __shfl_xor(ss, m, 32);
  if (lane == 0) dist[row] = sqrtf(ss);
}

// ---------------------------------------------------------------------------
// Kernel 6: deterministic per-cluster distance segment-sum.
// ---------------------------------------------------------------------------
__global__ __launch_bounds__(128) void k_dist_seg(
    const float* __restrict__ dist, const int* __restrict__ labels, int lvl,
    float* __restrict__ dist_sum) {
  __shared__ float part[128];
  const int c = blockIdx.x, t = threadIdx.x;
  float acc = 0.f;
  for (int i = t; i < B_ROWS; i += 128)
    if (labels[i * NLVL + lvl] == c) acc += dist[i];
  part[t] = acc;
  __syncthreads();
  for (int s = 64; s >= 1; s >>= 1) { if (t < s) part[t] += part[t + s]; __syncthreads(); }
  if (t == 0) dist_sum[c] = part[0];
}

// ---------------------------------------------------------------------------
// Kernel 7: densities.
// ---------------------------------------------------------------------------
__global__ __launch_bounds__(256) void k_dens(
    const float* __restrict__ sizes, const float* __restrict__ dist_sum,
    float* __restrict__ dens) {
  const int c = threadIdx.x;
  const float s    = sizes[c];
  const float safe = fmaxf(s, 1.0f);
  const float md   = dist_sum[c] / safe;
  const float d    = (s > 0.f) ? md / (safe * logf(s + ALPHA_C)) : 0.f;
  dens[c] = fmaxf(d, EPS_DENS);
}

// ---------------------------------------------------------------------------
// Kernel 8: proto loss via WMMA (8192x256 scaled logits), fused online
// softmax + gather-at-label. 1/dens is unbounded, so keep the online max
// here. dens cached in LDS.
// ---------------------------------------------------------------------------
__global__ __launch_bounds__(128) void k_proto_loss(
    const _Float16* __restrict__ feat_h, const _Float16* __restrict__ proto_h,
    const float* __restrict__ dens, const int* __restrict__ labels, int lvl,
    float* __restrict__ out) {
  __shared__ float sdens[NCLUS];
  __shared__ float red[4][16][3];
  const int wave = threadIdx.x >> 5;
  const int lane = threadIdx.x & 31;
  const int hi   = (lane >> 4) & 1;
  const int r0   = blockIdx.x * 16;

  for (int i = threadIdx.x; i < NCLUS; i += 128) sdens[i] = dens[i];
  __syncthreads();

  const v16h a0 = load_frag_a(feat_h, r0,  0, lane);
  const v16h a1 = load_frag_a(feat_h, r0, 32, lane);
  const v16h a2 = load_frag_a(feat_h, r0, 64, lane);
  const v16h a3 = load_frag_a(feat_h, r0, 96, lane);

  int labRow[8];
#pragma unroll
  for (int j = 0; j < 8; ++j)
    labRow[j] = labels[(r0 + j + hi * 8) * NLVL + lvl];

  float rmax[8], rsum[8], rpick[8];
#pragma unroll
  for (int j = 0; j < 8; ++j) { rmax[j] = NEG_BIG; rsum[j] = 0.f; rpick[j] = 0.f; }

  for (int nt = wave; nt < NCLUS / 16; nt += 4) {
    const int n0 = nt * 16;
    const v16h b0 = load_frag_b(proto_h, n0,  0, lane);
    const v16h b1 = load_frag_b(proto_h, n0, 32, lane);
    const v16h b2 = load_frag_b(proto_h, n0, 64, lane);
    const v16h b3 = load_frag_b(proto_h, n0, 96, lane);
    v8f acc = {};
    acc = WMMA_F16(a0, b0, acc);
    acc = WMMA_F16(a1, b1, acc);
    acc = WMMA_F16(a2, b2, acc);
    acc = WMMA_F16(a3, b3, acc);
    const int   c    = n0 + (lane & 15);
    const float invd = 1.0f / sdens[c];   // dens >= EPS_DENS
#pragma unroll
    for (int j = 0; j < 8; ++j) {
      const float v  = acc[j] * invd;
      const float mo = rmax[j];
      const float mn = fmaxf(mo, v);
      rsum[j] = rsum[j] * __expf(mo - mn) + __expf(v - mn);
      rmax[j] = mn;
      rpick[j] += (c == labRow[j]) ? v : 0.f;   // exactly one column matches
    }
  }
#pragma unroll
  for (int j = 0; j < 8; ++j) {
#pragma unroll
    for (int m = 1; m < 16; m <<= 1) {
      const float m2 = __shfl_xor(rmax[j], m, 32);
      const float s2 = __shfl_xor(rsum[j], m, 32);
      const float p2 = __shfl_xor(rpick[j], m, 32);
      const float mn = fmaxf(rmax[j], m2);
      rsum[j] = rsum[j] * __expf(rmax[j] - mn) + s2 * __expf(m2 - mn);
      rmax[j] = mn;
      rpick[j] += p2;
    }
  }
  if ((lane & 15) == 0) {
#pragma unroll
    for (int j = 0; j < 8; ++j) {
      const int m = j + hi * 8;
      red[wave][m][0] = rmax[j]; red[wave][m][1] = rsum[j]; red[wave][m][2] = rpick[j];
    }
  }
  __syncthreads();
  if (threadIdx.x < 16) {
    const int m = threadIdx.x;
    float gm = NEG_BIG;
    for (int w = 0; w < 4; ++w) gm = fmaxf(gm, red[w][m][0]);
    float gs = 0.f, gp = 0.f;
    for (int w = 0; w < 4; ++w) {
      gs += red[w][m][1] * __expf(red[w][m][0] - gm);
      gp += red[w][m][2];
    }
    const float lse = gm + logf(gs);
    out[r0 + m] += (lse - gp) / (float)NLVL;
  }
}

// ---------------------------------------------------------------------------
// Launch. Workspace layout (bytes), total ~6.6 MB:
//   [0)        feat_n  fp32 8192x128   (4,194,304)
//   [4194304)  feat_h  f16  8192x128   (2,097,152)
//   [6291456)  proto   fp32 256x128    (131,072)
//   [6422528)  proto_h f16  256x128    (65,536)
//   [6488064)  sizes   fp32 256
//   [6489088)  dist_sum fp32 256
//   [6490112)  dens    fp32 256
//   [6491136)  dist    fp32 8192       (32,768)
// ---------------------------------------------------------------------------
extern "C" void kernel_launch(void* const* d_in, const int* in_sizes, int n_in,
                              void* d_out, int out_size, void* d_ws, size_t ws_size,
                              hipStream_t stream) {
  const float* feat   = (const float*)d_in[0];
  const int*   labels = (const int*)d_in[1];
  float*       out    = (float*)d_out;
  char*        ws     = (char*)d_ws;

  float*    feat_n   = (float*)(ws + 0);
  _Float16* feat_h   = (_Float16*)(ws + 4194304);
  float*    proto    = (float*)(ws + 6291456);
  _Float16* proto_h  = (_Float16*)(ws + 6422528);
  float*    sizes    = (float*)(ws + 6488064);
  float*    dist_sum = (float*)(ws + 6489088);
  float*    dens     = (float*)(ws + 6490112);
  float*    dist     = (float*)(ws + 6491136);

  k_normalize<<<B_ROWS / 8, 256, 0, stream>>>(feat, feat_n, feat_h);
  k_insloss<<<B_ROWS / 16, 128, 0, stream>>>(feat_h, labels, out);

  for (int lvl = 0; lvl < NLVL; ++lvl) {
    k_proto_accum<<<NCLUS, 128, 0, stream>>>(feat_n, labels, lvl, proto, sizes);
    k_proto_final<<<NCLUS, 128, 0, stream>>>(proto, sizes, proto_h);
    k_dist_row<<<B_ROWS / 8, 256, 0, stream>>>(feat_n, proto, labels, lvl, dist);
    k_dist_seg<<<NCLUS, 128, 0, stream>>>(dist, labels, lvl, dist_sum);
    k_dens<<<1, 256, 0, stream>>>(sizes, dist_sum, dens);
    k_proto_loss<<<B_ROWS / 16, 128, 0, stream>>>(feat_h, proto_h, dens, labels, lvl, out);
  }
  (void)in_sizes; (void)n_in; (void)out_size; (void)ws_size;
}
